// DeformableTransformer_31851477467857
// MI455X (gfx1250) — compile-verified, matrix operand order checked by reading, not run on hardware
//
#include <hip/hip_runtime.h>
#include <hip/hip_bf16.h>

// ---------------------------------------------------------------------------
// Deformable transformer encoder for MI455X (gfx1250, wave32, WMMA).
//  * All dense projections -> v_wmma_f32_16x16x32_f16 (f32 accumulate).
//  * Weights pre-converted once per launch to f16, transposed [n][k], so the
//    GEMM B tile is staged global->LDS with GLOBAL_LOAD_ASYNC_TO_LDS_B128
//    (ASYNCcnt) -- no VGPR round trip, no per-workgroup re-conversion.
//  * A tile: float4 global load + packed ds_store_b64, f32->f16 in VGPRs.
//  * LayerNorm fused into GEMM epilogue (workgroup owns full 256-wide rows).
//  * Sampling: wave = (b,s,head), lane = channel; 4-pt softmax + 16 gathers.
// ---------------------------------------------------------------------------

#define D_       256
#define HEADS_   8
#define POINTS_  4
#define LAYERS_  6
#define HGT_     100
#define WID_     100
#define DH_      32
#define S_       (HGT_ * WID_)
#define BS_      2
#define M_       (BS_ * S_)        // 20000 rows

typedef __attribute__((ext_vector_type(16))) _Float16 v16h;
typedef __attribute__((ext_vector_type(8)))  _Float16 v8h;
typedef __attribute__((ext_vector_type(4)))  _Float16 v4h;
typedef __attribute__((ext_vector_type(4)))  float    v4f;
typedef __attribute__((ext_vector_type(8)))  float    v8f;
typedef __attribute__((ext_vector_type(4)))  int      v4i;

// 16-byte global(f16)->LDS copy. Prefer the CDNA5 async DMA path (ASYNCcnt).
__device__ __forceinline__ void async_copy16(_Float16* lds, const _Float16* g) {
#if __has_builtin(__builtin_amdgcn_global_load_async_to_lds_b128)
    typedef __attribute__((address_space(1))) v4i gv4i;
    typedef __attribute__((address_space(3))) v4i lv4i;
    __builtin_amdgcn_global_load_async_to_lds_b128(
        (gv4i*)(void*)g, (lv4i*)(void*)lds, 0, 0);
#else
    *(v8h*)lds = *(const v8h*)g;
#endif
}
__device__ __forceinline__ void wait_async() {
#if __has_builtin(__builtin_amdgcn_global_load_async_to_lds_b128)
  #if __has_builtin(__builtin_amdgcn_s_wait_asynccnt)
    __builtin_amdgcn_s_wait_asynccnt(0);
  #else
    asm volatile("s_wait_asynccnt 0" ::: "memory");
  #endif
#endif
}

// WMMA operand fragment from LDS row staged with 40-half (80 B) stride.
// ISA 7.12.2 (16-bit 16x32): lane<16: halves 0..7=K0..7, 8..15=K16..23;
// lane>=16: halves 0..7=K8..15, 8..15=K24..31  -> two ds_load_b128.
__device__ __forceinline__ v16h load_frag(const _Float16* base, int laneHalf) {
    union { v16h v; v8h h[2]; } u;
    u.h[0] = *(const v8h*)(base + laneHalf * 8);
    u.h[1] = *(const v8h*)(base + 16 + laneHalf * 8);
    return u.v;
}

// ---------------------------------------------------------------------------
// Weight prep: WT[n*256 + k] = (f16) W[k*N + n]   (W is [256, N] row-major)
// ---------------------------------------------------------------------------
__global__ __launch_bounds__(256) void prep_wT_kernel(
    const float* __restrict__ W, _Float16* __restrict__ WT, int N)
{
    int e = blockIdx.x * 256 + threadIdx.x;
    if (e >= 256 * N) return;
    int k = e / N, n = e % N;
    WT[(size_t)n * 256 + k] = (_Float16)W[(size_t)k * N + n];
}

// ---------------------------------------------------------------------------
// gemm256<EPI>: out[M,256] = epilogue(A[M,256] @ B[256,256] + bias)
//   EPI=0: + bias      EPI=1: + bias, ReLU
//   EPI=2: + bias + residual, then row LayerNorm(gamma,beta)
// Block: 256 threads (8 waves). M-tile = 32 rows, N = full 256.
// BT is f16, transposed [n][k].
// ---------------------------------------------------------------------------
template <int EPI>
__global__ __launch_bounds__(256) void gemm256_kernel(
    const float* __restrict__ A, const _Float16* __restrict__ BT,
    const float* __restrict__ bias, const float* __restrict__ resid,
    const float* __restrict__ gamma, const float* __restrict__ beta,
    float* __restrict__ out, int M)
{
    __shared__ alignas(16) _Float16 lA[32 * 40];
    __shared__ alignas(16) _Float16 lB[256 * 40];
    __shared__ float tile[(EPI == 2) ? 32 * 264 : 1];

    const int tid      = threadIdx.x;
    const int wave     = tid >> 5;
    const int lane     = tid & 31;
    const int laneLo   = lane & 15;
    const int laneHalf = lane >> 4;
    const int rowBase  = blockIdx.x * 32;
    const int nBase    = wave * 32;
    const int ar       = tid >> 3;        // A stage: row 0..31
    const int ak       = (tid & 7) * 4;   // A stage: k quarter

    v8f acc[2][2] = {};

    for (int k0 = 0; k0 < 256; k0 += 32) {
        // Stage B: 4 x async b128 per thread, row n = tid (pure f16 copy).
        {
            const _Float16* g = BT + (size_t)tid * 256 + k0;
            _Float16* l = &lB[tid * 40];
            async_copy16(l,      g);
            async_copy16(l + 8,  g + 8);
            async_copy16(l + 16, g + 16);
            async_copy16(l + 24, g + 24);
        }
        // Stage A: float4 load, f32->f16, ds_store_b64.
        {
            int row = rowBase + ar;
            v4f av = {0.f, 0.f, 0.f, 0.f};
            if (row < M) av = *(const v4f*)&A[(size_t)row * 256 + k0 + ak];
            v4h hv = { (_Float16)av.x, (_Float16)av.y,
                       (_Float16)av.z, (_Float16)av.w };
            *(v4h*)&lA[ar * 40 + ak] = hv;
        }
        wait_async();
        __syncthreads();

        v16h a0 = load_frag(&lA[(laneLo) * 40], laneHalf);
        v16h a1 = load_frag(&lA[(16 + laneLo) * 40], laneHalf);
        v16h b0 = load_frag(&lB[(nBase + laneLo) * 40], laneHalf);
        v16h b1 = load_frag(&lB[(nBase + 16 + laneLo) * 40], laneHalf);
        acc[0][0] = __builtin_amdgcn_wmma_f32_16x16x32_f16(false, a0, false, b0, (short)0, acc[0][0], false, false);
        acc[0][1] = __builtin_amdgcn_wmma_f32_16x16x32_f16(false, a0, false, b1, (short)0, acc[0][1], false, false);
        acc[1][0] = __builtin_amdgcn_wmma_f32_16x16x32_f16(false, a1, false, b0, (short)0, acc[1][0], false, false);
        acc[1][1] = __builtin_amdgcn_wmma_f32_16x16x32_f16(false, a1, false, b1, (short)0, acc[1][1], false, false);
        __syncthreads();
    }

    if constexpr (EPI == 2) {
        __shared__ float pSum[32][8], pSq[32][8], sMean[32], sRstd[32];
        // D layout: element r of lane L -> row r + 8*(L>>4), col (L&15).
        #pragma unroll
        for (int ms = 0; ms < 2; ++ms)
        #pragma unroll
        for (int ns = 0; ns < 2; ++ns) {
            int col = nBase + ns * 16 + laneLo;
            float bv = bias[col];
            #pragma unroll
            for (int r = 0; r < 8; ++r) {
                int mRow = ms * 16 + r + 8 * laneHalf;
                int row  = rowBase + mRow;
                float v = acc[ms][ns][r] + bv;
                if (row < M) v += resid[(size_t)row * 256 + col];
                tile[mRow * 264 + col] = v;
            }
        }
        __syncthreads();
        {   // 8 threads per row: partial sum / sumsq over 32 columns.
            int r32 = tid >> 3, seg = tid & 7;
            float s = 0.f, sq = 0.f;
            #pragma unroll
            for (int j = 0; j < 32; ++j) {
                float v = tile[r32 * 264 + seg * 32 + j];
                s += v; sq += v * v;
            }
            pSum[r32][seg] = s; pSq[r32][seg] = sq;
        }
        __syncthreads();
        if (tid < 32) {
            float s = 0.f, sq = 0.f;
            #pragma unroll
            for (int j = 0; j < 8; ++j) { s += pSum[tid][j]; sq += pSq[tid][j]; }
            float mean = s * (1.f / 256.f);
            float var  = sq * (1.f / 256.f) - mean * mean;
            sMean[tid] = mean;
            sRstd[tid] = rsqrtf(var + 1e-5f);
        }
        __syncthreads();
        float g = gamma[tid], bb = beta[tid];
        #pragma unroll
        for (int i = 0; i < 32; ++i) {
            int row = rowBase + i;
            if (row < M)
                out[(size_t)row * 256 + tid] =
                    (tile[i * 264 + tid] - sMean[i]) * sRstd[i] * g + bb;
        }
    } else {
        #pragma unroll
        for (int ms = 0; ms < 2; ++ms)
        #pragma unroll
        for (int ns = 0; ns < 2; ++ns) {
            int col = nBase + ns * 16 + laneLo;
            float bv = bias[col];
            #pragma unroll
            for (int r = 0; r < 8; ++r) {
                int row = rowBase + ms * 16 + r + 8 * laneHalf;
                float v = acc[ms][ns][r] + bv;
                if (EPI == 1) v = v > 0.f ? v : 0.f;
                if (row < M) out[(size_t)row * 256 + col] = v;
            }
        }
    }
}

// ---------------------------------------------------------------------------
// qproj: q = src + pos; off = q @ off_w + off_b (cols 0..63)
//                       aw  = q @ aw_w  + aw_b  (cols 64..95)
// qT is f16 transposed [96][256]. Block: 192 threads (6 waves), M-tile 64,
// each wave owns one 16-wide N tile.
// ---------------------------------------------------------------------------
__global__ __launch_bounds__(192) void qproj_kernel(
    const float* __restrict__ src, const float* __restrict__ pos,
    const _Float16* __restrict__ qT,
    const float* __restrict__ offB, const float* __restrict__ awB,
    float* __restrict__ offOut, float* __restrict__ awOut, int M)
{
    __shared__ alignas(16) _Float16 lA[64 * 40];
    __shared__ alignas(16) _Float16 lB[96 * 40];
    const int tid      = threadIdx.x;
    const int wave     = tid >> 5;
    const int lane     = tid & 31;
    const int laneLo   = lane & 15;
    const int laneHalf = lane >> 4;
    const int rowBase  = blockIdx.x * 64;

    v8f acc[4] = {};
    for (int k0 = 0; k0 < 256; k0 += 32) {
        for (int e = tid; e < 96 * 4; e += 192) {   // B: async b128 copies
            int n = e >> 2, part = e & 3;
            async_copy16(&lB[n * 40 + part * 8],
                         qT + (size_t)n * 256 + k0 + part * 8);
        }
        for (int e = tid; e < 64 * 8; e += 192) {   // A: (src+pos) -> f16
            int r = e >> 3, k4 = (e & 7) * 4;
            int row = rowBase + r;
            v4f s = {0.f, 0.f, 0.f, 0.f}, p = {0.f, 0.f, 0.f, 0.f};
            if (row < M) {
                s = *(const v4f*)&src[(size_t)row * 256 + k0 + k4];
                p = *(const v4f*)&pos[(size_t)row * 256 + k0 + k4];
            }
            v4h hv = { (_Float16)(s.x + p.x), (_Float16)(s.y + p.y),
                       (_Float16)(s.z + p.z), (_Float16)(s.w + p.w) };
            *(v4h*)&lA[r * 40 + k4] = hv;
        }
        wait_async();
        __syncthreads();
        v16h b = load_frag(&lB[(wave * 16 + laneLo) * 40], laneHalf);
        #pragma unroll
        for (int ms = 0; ms < 4; ++ms) {
            v16h a = load_frag(&lA[(ms * 16 + laneLo) * 40], laneHalf);
            acc[ms] = __builtin_amdgcn_wmma_f32_16x16x32_f16(false, a, false, b, (short)0, acc[ms], false, false);
        }
        __syncthreads();
    }
    int n = wave * 16 + laneLo;
    float bv = (n < 64) ? offB[n] : awB[n - 64];
    #pragma unroll
    for (int ms = 0; ms < 4; ++ms)
    #pragma unroll
    for (int r = 0; r < 8; ++r) {
        int row = rowBase + ms * 16 + r + 8 * laneHalf;
        if (row < M) {
            float v = acc[ms][r] + bv;
            if (n < 64) offOut[(size_t)row * 64 + n] = v;
            else        awOut[(size_t)row * 32 + (n - 64)] = v;
        }
    }
}

// ---------------------------------------------------------------------------
// MSDeformAttn sampling: one wave per (b, s, head), lane = channel (DH=32).
// ---------------------------------------------------------------------------
__global__ __launch_bounds__(256) void msda_sample_kernel(
    const float* __restrict__ value, const float* __restrict__ off,
    const float* __restrict__ aw, float* __restrict__ attn)
{
    const int tid = threadIdx.x;
    const int gw  = blockIdx.x * 8 + (tid >> 5);
    const int d   = tid & 31;
    if (gw >= BS_ * S_ * HEADS_) return;
    const int h   = gw % HEADS_;
    const int s   = (gw / HEADS_) % S_;
    const int b   = gw / (HEADS_ * S_);
    const int row = b * S_ + s;

    float l0 = aw[(size_t)row * 32 + h * 4 + 0];
    float l1 = aw[(size_t)row * 32 + h * 4 + 1];
    float l2 = aw[(size_t)row * 32 + h * 4 + 2];
    float l3 = aw[(size_t)row * 32 + h * 4 + 3];
    float mx = fmaxf(fmaxf(l0, l1), fmaxf(l2, l3));
    float e0 = __expf(l0 - mx), e1 = __expf(l1 - mx);
    float e2 = __expf(l2 - mx), e3 = __expf(l3 - mx);
    float inv = 1.f / (e0 + e1 + e2 + e3);
    float wp[4] = { e0 * inv, e1 * inv, e2 * inv, e3 * inv };

    const float rx = ((s % WID_) + 0.5f) / WID_;
    const float ry = ((s / WID_) + 0.5f) / HGT_;

    float accv = 0.f;
    #pragma unroll
    for (int p = 0; p < POINTS_; ++p) {
        float ox = off[(size_t)row * 64 + h * 8 + p * 2 + 0];
        float oy = off[(size_t)row * 64 + h * 8 + p * 2 + 1];
        float x = (rx + ox / WID_) * WID_ - 0.5f;
        float y = (ry + oy / HGT_) * HGT_ - 0.5f;
        int   x0 = (int)floorf(x), y0 = (int)floorf(y);
        float wx1 = x - x0, wx0 = 1.f - wx1;
        float wy1 = y - y0, wy0 = 1.f - wy1;
        #pragma unroll
        for (int c = 0; c < 4; ++c) {
            int   xi = x0 + (c & 1), yi = y0 + (c >> 1);
            float wc = ((c & 1) ? wx1 : wx0) * ((c >> 1) ? wy1 : wy0);
            bool  valid = (xi >= 0) & (xi < WID_) & (yi >= 0) & (yi < HGT_);
            int   xc = min(max(xi, 0), WID_ - 1);
            int   yc = min(max(yi, 0), HGT_ - 1);
            int   idx = yc * WID_ + xc;
            float v = value[((size_t)(b * S_ + idx)) * 256 + h * 32 + d];
            accv += v * (wc * (valid ? wp[p] : 0.f));
        }
    }
    attn[(size_t)row * 256 + h * 32 + d] = accv;
}

// ---------------------------------------------------------------------------
// init: src[b,s,c] = features[b,c,s]; pos[b,s,c] = pos_embed[b,c,s] + lvl[c]
// ---------------------------------------------------------------------------
__global__ __launch_bounds__(256) void init_kernel(
    const float* __restrict__ features, const float* __restrict__ pos_embed,
    const float* __restrict__ level_embed,
    float* __restrict__ src, float* __restrict__ pos)
{
    int e = blockIdx.x * 256 + threadIdx.x;
    if (e >= BS_ * S_ * D_) return;
    int c = e & 255;
    int s = (e >> 8) % S_;
    int b = e / (S_ * D_);
    size_t fidx = ((size_t)b * D_ + c) * S_ + s;
    src[e] = features[fidx];
    pos[e] = pos_embed[fidx] + level_embed[c];
}

// ---------------------------------------------------------------------------
extern "C" void kernel_launch(void* const* d_in, const int* in_sizes, int n_in,
                              void* d_out, int out_size, void* d_ws, size_t ws_size,
                              hipStream_t stream)
{
    (void)in_sizes; (void)n_in; (void)out_size; (void)ws_size;
    const float* features    = (const float*)d_in[0];
    const float* pos_embed   = (const float*)d_in[1];
    const float* level_embed = (const float*)d_in[2];
    const float* off_w  = (const float*)d_in[3];
    const float* off_b  = (const float*)d_in[4];
    const float* aw_w   = (const float*)d_in[5];
    const float* aw_b   = (const float*)d_in[6];
    const float* val_w  = (const float*)d_in[7];
    const float* val_b  = (const float*)d_in[8];
    const float* out_w  = (const float*)d_in[9];
    const float* out_b  = (const float*)d_in[10];
    const float* ln1_g  = (const float*)d_in[11];
    const float* ln1_b  = (const float*)d_in[12];
    const float* ffn_w1 = (const float*)d_in[13];
    const float* ffn_b1 = (const float*)d_in[14];
    const float* ffn_w2 = (const float*)d_in[15];
    const float* ffn_b2 = (const float*)d_in[16];
    const float* ln3_g  = (const float*)d_in[17];
    const float* ln3_b  = (const float*)d_in[18];

    float* ws = (float*)d_ws;
    const size_t NSD = (size_t)BS_ * S_ * D_;     // 5,120,000 floats
    float* src   = ws;                            // [M,256]
    float* pos   = src + NSD;                     // [M,256]
    float* value = pos + NSD;                     // [M,256]
    float* attn  = value + NSD;                   // [M,256] (also ffn hidden)
    float* offb  = attn + NSD;                    // [M,64]
    float* awb   = offb + (size_t)M_ * 64;        // [M,32]

    // f16 transposed weights (per layer: valT,outT,f1T,f2T [256][256], qT [96][256])
    _Float16* wt16 = (_Float16*)(awb + (size_t)M_ * 32);
    const size_t W256 = 256 * 256;
    const size_t LSTRIDE = 4 * W256 + 96 * 256;   // 286720 halfs / layer

    // One-time (per launch) weight prep.
    for (int l = 0; l < LAYERS_; ++l) {
        _Float16* valT = wt16 + (size_t)l * LSTRIDE;
        _Float16* outT = valT + W256;
        _Float16* f1T  = outT + W256;
        _Float16* f2T  = f1T + W256;
        _Float16* qT   = f2T + W256;
        prep_wT_kernel<<<(256 * 256 + 255) / 256, 256, 0, stream>>>(val_w  + (size_t)l * W256, valT, 256);
        prep_wT_kernel<<<(256 * 256 + 255) / 256, 256, 0, stream>>>(out_w  + (size_t)l * W256, outT, 256);
        prep_wT_kernel<<<(256 * 256 + 255) / 256, 256, 0, stream>>>(ffn_w1 + (size_t)l * W256, f1T, 256);
        prep_wT_kernel<<<(256 * 256 + 255) / 256, 256, 0, stream>>>(ffn_w2 + (size_t)l * W256, f2T, 256);
        prep_wT_kernel<<<(256 * 64 + 255) / 256, 256, 0, stream>>>(off_w + (size_t)l * 256 * 64, qT, 64);
        prep_wT_kernel<<<(256 * 32 + 255) / 256, 256, 0, stream>>>(aw_w  + (size_t)l * 256 * 32, qT + 64 * 256, 32);
    }

    init_kernel<<<(int)((NSD + 255) / 256), 256, 0, stream>>>(
        features, pos_embed, level_embed, src, pos);

    const int gemmGrid = (M_ + 31) / 32;                 // 625
    const int qGrid    = (M_ + 63) / 64;                 // 313
    const int sampGrid = (BS_ * S_ * HEADS_) / 8;        // 20000

    for (int l = 0; l < LAYERS_; ++l) {
        _Float16* valT = wt16 + (size_t)l * LSTRIDE;
        _Float16* outT = valT + W256;
        _Float16* f1T  = outT + W256;
        _Float16* f2T  = f1T + W256;
        _Float16* qT   = f2T + W256;
        // value = src @ val_w + val_b
        gemm256_kernel<0><<<gemmGrid, 256, 0, stream>>>(
            src, valT, val_b + l * D_, nullptr, nullptr, nullptr, value, M_);
        // off/aw logits from q = src + pos
        qproj_kernel<<<qGrid, 192, 0, stream>>>(
            src, pos, qT, off_b + l * 64, aw_b + l * 32, offb, awb, M_);
        // deformable sampling -> attn
        msda_sample_kernel<<<sampGrid, 256, 0, stream>>>(value, offb, awb, attn);
        // src = LN1(src + attn @ out_w + out_b)   (in place, rows disjoint)
        gemm256_kernel<2><<<gemmGrid, 256, 0, stream>>>(
            attn, outT, out_b + l * D_, src, ln1_g + l * D_, ln1_b + l * D_, src, M_);
        // hidden = relu(src @ ffn_w1 + ffn_b1)  -> reuse attn buffer
        gemm256_kernel<1><<<gemmGrid, 256, 0, stream>>>(
            src, f1T, ffn_b1 + l * D_, nullptr, nullptr, nullptr, attn, M_);
        // src = LN3(src + hidden @ ffn_w2 + ffn_b2); last layer -> d_out
        float* dst = (l == LAYERS_ - 1) ? (float*)d_out : src;
        gemm256_kernel<2><<<gemmGrid, 256, 0, stream>>>(
            attn, f2T, ffn_b2 + l * D_, src, ln3_g + l * D_, ln3_b + l * D_, dst, M_);
    }
}
